// bimpm_2508260901186
// MI455X (gfx1250) — matrix operand order repaired and checked by reading, not compiled, not run on hardware
//
#include <hip/hip_runtime.h>
#include <math.h>

#define EPS_F 1e-8f

typedef __attribute__((ext_vector_type(16))) _Float16 v16h;
typedef __attribute__((ext_vector_type(8)))  float    v8f;

// problem dims (fixed by reference)
#define BB   16
#define LLN  128
#define EE   300
#define EPAD 320
#define HHN  128
#define GGN  512      // 4*H
#define PPN  4
#define BLN  (BB*LLN)
#define MVW  32       // 8*P

__device__ __forceinline__ float sigf(float x){ return 1.f/(1.f+__expf(-x)); }
__device__ __forceinline__ float divsv(float n, float d){ return n / (d > EPS_F ? d : EPS_F); }

// ---------------- WMMA fragment helpers (wave32, v_wmma_f32_16x16x32_f16) ----
// A: 16x32 tile, row-major f16 source, lda in elements.
// Layout: lanes 0-15 -> M=lane, K=0..15; lanes 16-31 -> M=lane-16, K=16..31.
__device__ __forceinline__ v16h frag_a(const _Float16* A, int lda) {
  int lane = threadIdx.x & 31;
  int m  = lane & 15;
  int kh = (lane >> 4) << 4;          // 0 or 16
  const _Float16* r = A + m*lda;
  v16h a;
#pragma unroll
  for (int j=0;j<16;++j){
    a[j] = r[kh + j];
  }
  return a;
}
// B operand for C = A * Bw^T, Bw row-major (N,K): element(k,n) = Bw[n*ldb + k].
// Lanes 0-15 -> N=lane, K=0..15; lanes 16-31 -> N=lane-16, K=16..31.
__device__ __forceinline__ v16h frag_b_nt(const _Float16* Bw, int ldb) {
  int lane = threadIdx.x & 31;
  int n  = lane & 15;
  int kb = (lane >> 4) << 4;          // 0 or 16
  const _Float16* r = Bw + n*ldb + kb;
  v16h b;
#pragma unroll
  for (int j=0;j<16;++j) b[j] = r[j];
  return b;
}
// B operand for C = A * Bm, Bm row-major (K,N): element(k,n) = Bm[k*ldb + n].
__device__ __forceinline__ v16h frag_b_nn(const _Float16* Bm, int ldb) {
  int lane = threadIdx.x & 31;
  int n  = lane & 15;
  int kb = (lane >> 4) << 4;
  v16h b;
#pragma unroll
  for (int j=0;j<16;++j) b[j] = Bm[(kb+j)*ldb + n];
  return b;
}
// C/D: VGPR r -> (M = r + 8*(lane>=16), N = lane&15)
__device__ __forceinline__ void frag_c_store(float* C, int ldc, v8f c) {
  int lane = threadIdx.x & 31;
  int n  = lane & 15;
  int mo = (lane >> 4) << 3;
#pragma unroll
  for (int r=0;r<8;++r) C[(mo+r)*ldc + n] = c[r];
}

// ---------------- generic batched WMMA GEMMs --------------------------------
// C(M,N) = A(M,K) * Bw(N,K)^T ; all f16 in, f32 out; K % 32 == 0, M,N % 16 == 0
__global__ __launch_bounds__(128) void bimpm_gemm_nt(
    const _Float16* __restrict__ A, const _Float16* __restrict__ Bw,
    float* __restrict__ C, int M, int N, int K,
    long long sA, long long sB, long long sC) {
  A  += (long long)blockIdx.y * sA;
  Bw += (long long)blockIdx.y * sB;
  C  += (long long)blockIdx.y * sC;
  int tilesN = N >> 4;
  int tile = blockIdx.x * (blockDim.x >> 5) + (threadIdx.x >> 5);
  if (tile >= (M >> 4) * tilesN) return;
  int tm = tile / tilesN, tn = tile % tilesN;
  const _Float16* Ab = A  + (long long)tm * 16 * K;
  const _Float16* Bb = Bw + (long long)tn * 16 * K;
  v8f acc = {};
  for (int k = 0; k < K; k += 32)
    acc = __builtin_amdgcn_wmma_f32_16x16x32_f16(
        false, frag_a(Ab + k, K), false, frag_b_nt(Bb + k, K),
        (short)0, acc, false, false);
  frag_c_store(C + (long long)tm * 16 * N + tn * 16, N, acc);
}

// C(M,N) = A(M,K) * Bm(K,N)
__global__ __launch_bounds__(128) void bimpm_gemm_nn(
    const _Float16* __restrict__ A, const _Float16* __restrict__ Bm,
    float* __restrict__ C, int M, int N, int K,
    long long sA, long long sB, long long sC) {
  A  += (long long)blockIdx.y * sA;
  Bm += (long long)blockIdx.y * sB;
  C  += (long long)blockIdx.y * sC;
  int tilesN = N >> 4;
  int tile = blockIdx.x * (blockDim.x >> 5) + (threadIdx.x >> 5);
  if (tile >= (M >> 4) * tilesN) return;
  int tm = tile / tilesN, tn = tile % tilesN;
  const _Float16* Ab = A + (long long)tm * 16 * K;
  v8f acc = {};
  for (int k = 0; k < K; k += 32)
    acc = __builtin_amdgcn_wmma_f32_16x16x32_f16(
        false, frag_a(Ab + k, K), false,
        frag_b_nn(Bm + (long long)k * N + tn * 16, N),
        (short)0, acc, false, false);
  frag_c_store(C + (long long)tm * 16 * N + tn * 16, N, acc);
}

// ---------------- data movement / conversion --------------------------------
__global__ void bimpm_gather_embed(const int* __restrict__ tok,
                                   const float* __restrict__ emb,
                                   _Float16* __restrict__ x16, int total) {
  int idx = blockIdx.x * blockDim.x + threadIdx.x;
  if (idx >= total) return;
  int e = idx % EPAD, r = idx / EPAD;
  x16[idx] = (e < EE) ? (_Float16)emb[(long long)tok[r] * EE + e] : (_Float16)0.f;
}

__global__ void bimpm_convert_pad(const float* __restrict__ src,
                                  _Float16* __restrict__ dst,
                                  int rows, int K, int Kpad) {
  int idx = blockIdx.x * blockDim.x + threadIdx.x;
  if (idx >= rows * Kpad) return;
  int kp = idx % Kpad, r = idx / Kpad;
  dst[idx] = (kp < K) ? (_Float16)src[(long long)r * K + kp] : (_Float16)0.f;
}

__global__ void bimpm_f32_to_f16(const float* __restrict__ s,
                                 _Float16* __restrict__ d, int n) {
  int i = blockIdx.x * blockDim.x + threadIdx.x;
  if (i < n) d[i] = (_Float16)s[i];
}

__global__ void bimpm_add_bias2(float* __restrict__ xg,
                                const float* __restrict__ b1,
                                const float* __restrict__ b2, int total) {
  int i = blockIdx.x * blockDim.x + threadIdx.x;
  if (i >= total) return;
  int g = i % GGN;
  xg[i] += b1[g] + b2[g];
}

// ---------------- LSTM scan: 4 independent scans, 1 block each --------------
// blocks 0,1: forward (WhhF); blocks 2,3: reverse-time (WhhB).
// xg is always stored in natural time order; reverse scans index it reversed
// and write hs at the reversed position (== un-reversed output).
__global__ __launch_bounds__(256) void bimpm_lstm_scan4(
    const float* xgA, const float* xgB, const float* xgC, const float* xgD,
    const _Float16* __restrict__ WhhF, const _Float16* __restrict__ WhhB,
    float* hsA, float* hsB, float* hsC, float* hsD,
    float* hTA, float* hTB, float* hTC, float* hTD) {
  __shared__ float    g_sh[BB][GGN];   // 32 KB gate pre-activations
  __shared__ _Float16 h16s[BB][HHN];   // 4 KB  hidden state (f16 for WMMA A)

  int blk = blockIdx.x;
  const float* xg = (blk==0)?xgA:(blk==1)?xgB:(blk==2)?xgC:xgD;
  const _Float16* Whh16 = (blk < 2) ? WhhF : WhhB;
  float* hs = (blk==0)?hsA:(blk==1)?hsB:(blk==2)?hsC:hsD;
  float* hT = (blk==0)?hTA:(blk==1)?hTB:(blk==2)?hTC:hTD;
  int rev = (blk >= 2);

  int tid  = threadIdx.x;
  int wave = tid >> 5;
  int lane = tid & 31;
  int b_own = tid & 15;            // batch row owned in gate phase
  int hh0   = (tid >> 4) * 8;      // 8 hidden units per thread

  float c_reg[8], h_reg[8];
#pragma unroll
  for (int j=0;j<8;++j){ c_reg[j]=0.f; h_reg[j]=0.f; h16s[b_own][hh0+j]=(_Float16)0.f; }
  __syncthreads();

  for (int t = 0; t < LLN; ++t) {
    int ts = rev ? (LLN - 1 - t) : t;
    // ---- recurrent matmul: g = h(16x128) @ Whh^T(128x512), 8 waves x 4 tiles
#pragma unroll
    for (int nt = 0; nt < 4; ++nt) {
      int ntile = wave * 4 + nt;
      v8f acc = {};
#pragma unroll
      for (int k = 0; k < HHN; k += 32)
        acc = __builtin_amdgcn_wmma_f32_16x16x32_f16(
            false, frag_a(&h16s[0][0] + k, HHN),
            false, frag_b_nt(Whh16 + (long long)ntile * 16 * HHN + k, HHN),
            (short)0, acc, false, false);
      int n  = lane & 15;
      int mo = (lane >> 4) << 3;
#pragma unroll
      for (int r = 0; r < 8; ++r)
        g_sh[mo + r][ntile * 16 + n] = acc[r];
    }
    __syncthreads();
    // ---- gates + state update (i,f,g,o order)
    const float* xr = xg + ((long long)(b_own * LLN + ts)) * GGN;
#pragma unroll
    for (int j = 0; j < 8; ++j) {
      int hh = hh0 + j;
      float gi = g_sh[b_own][hh]            + xr[hh];
      float gf = g_sh[b_own][HHN + hh]      + xr[HHN + hh];
      float gg = g_sh[b_own][2*HHN + hh]    + xr[2*HHN + hh];
      float go = g_sh[b_own][3*HHN + hh]    + xr[3*HHN + hh];
      float c  = sigf(gf) * c_reg[j] + sigf(gi) * tanhf(gg);
      float h  = sigf(go) * tanhf(c);
      c_reg[j] = c; h_reg[j] = h;
      h16s[b_own][hh] = (_Float16)h;
      hs[((long long)(b_own * LLN + ts)) * HHN + hh] = h;
    }
    __syncthreads();
  }
#pragma unroll
  for (int j = 0; j < 8; ++j) hT[b_own * HHN + hh0 + j] = h_reg[j];
}

// ---------------- norms / attention / matching ------------------------------
__global__ void bimpm_norm(const float* __restrict__ v, float* __restrict__ nrm) {
  int i = blockIdx.x * blockDim.x + threadIdx.x;   // (b*L + l)
  if (i >= BLN) return;
  const float* r = v + (long long)i * HHN;
  float a = 0.f;
  for (int h = 0; h < HHN; ++h) a += r[h] * r[h];
  nrm[i] = sqrtf(a);
}

__global__ void bimpm_wnorm(const float* __restrict__ v, const float* __restrict__ w,
                            float* __restrict__ wn) {
  int i = blockIdx.x * blockDim.x + threadIdx.x;   // (b*P + p)*L + l
  if (i >= BB * PPN * LLN) return;
  int l = i % LLN, p = (i / LLN) % PPN, b = i / (LLN * PPN);
  const float* r  = v + (long long)(b * LLN + l) * HHN;
  const float* wp = w + p * HHN;
  float a = 0.f;
  for (int h = 0; h < HHN; ++h) { float t = r[h] * wp[h]; a += t * t; }
  wn[i] = sqrtf(a);
}

__global__ void bimpm_scale_w_f16(const float* __restrict__ v, const float* __restrict__ w,
                                  _Float16* __restrict__ out) {
  int i = blockIdx.x * blockDim.x + threadIdx.x;   // ((b*P+p)*L+l)*H+h
  if (i >= BB * PPN * LLN * HHN) return;
  int h = i % HHN, l = (i / HHN) % LLN, p = (i / (HHN * LLN)) % PPN, b = i / (HHN * LLN * PPN);
  out[i] = (_Float16)(v[(long long)(b * LLN + l) * HHN + h] * w[p * HHN + h]);
}

__global__ void bimpm_att_normalize(float* __restrict__ S, const float* __restrict__ n1,
                                    const float* __restrict__ n2) {
  int i = blockIdx.x * blockDim.x + threadIdx.x;   // (b*L+l)*L+m
  if (i >= BB * LLN * LLN) return;
  int m = i % LLN, l = (i / LLN) % LLN, b = i / (LLN * LLN);
  S[i] = divsv(S[i], n1[b * LLN + l] * n2[b * LLN + m]);
}

__global__ void bimpm_rowsum(const float* __restrict__ att, float* __restrict__ s) {
  int i = blockIdx.x * blockDim.x + threadIdx.x;   // b*L + l
  if (i >= BLN) return;
  const float* r = att + (long long)i * LLN;
  float a = 0.f;
  for (int m = 0; m < LLN; ++m) a += r[m];
  s[i] = a;
}

__global__ void bimpm_colsum(const float* __restrict__ att, float* __restrict__ s) {
  int i = blockIdx.x * blockDim.x + threadIdx.x;   // b*L + m
  if (i >= BLN) return;
  int m = i % LLN, b = i / LLN;
  float a = 0.f;
  for (int l = 0; l < LLN; ++l) a += att[((long long)(b * LLN + l)) * LLN + m];
  s[i] = a;
}

__global__ void bimpm_att_f16_tr(const float* __restrict__ S, _Float16* __restrict__ a,
                                 _Float16* __restrict__ aT) {
  int i = blockIdx.x * blockDim.x + threadIdx.x;
  if (i >= BB * LLN * LLN) return;
  int m = i % LLN, l = (i / LLN) % LLN, b = i / (LLN * LLN);
  float v = S[i];
  a [((long long)(b * LLN + l)) * LLN + m] = (_Float16)v;
  aT[((long long)(b * LLN + m)) * LLN + l] = (_Float16)v;
}

__global__ void bimpm_div_rows(float* __restrict__ v, const float* __restrict__ s, int total) {
  int i = blockIdx.x * blockDim.x + threadIdx.x;
  if (i >= total) return;
  v[i] = divsv(v[i], s[i / HHN]);
}

// att_max over m: out[b,l,h] = max_m conh[b,m,h] * att[b,l,m]
__global__ void bimpm_attmax_h(const float* __restrict__ att, const float* __restrict__ conh,
                               float* __restrict__ out) {
  int i = blockIdx.x * blockDim.x + threadIdx.x;
  if (i >= BB * LLN * HHN) return;
  int h = i % HHN, l = (i / HHN) % LLN, b = i / (HHN * LLN);
  float best = -3.4e38f;
  for (int m = 0; m < LLN; ++m) {
    float v = conh[((long long)(b * LLN + m)) * HHN + h] * att[((long long)(b * LLN + l)) * LLN + m];
    best = fmaxf(best, v);
  }
  out[i] = best;
}
// att_max over l: out[b,m,h] = max_l conp[b,l,h] * att[b,l,m]
__global__ void bimpm_attmax_p(const float* __restrict__ att, const float* __restrict__ conp,
                               float* __restrict__ out) {
  int i = blockIdx.x * blockDim.x + threadIdx.x;
  if (i >= BB * LLN * HHN) return;
  int h = i % HHN, m = (i / HHN) % LLN, b = i / (HHN * LLN);
  float best = -3.4e38f;
  for (int l = 0; l < LLN; ++l) {
    float v = conp[((long long)(b * LLN + l)) * HHN + h] * att[((long long)(b * LLN + l)) * LLN + m];
    best = fmaxf(best, v);
  }
  out[i] = best;
}

// full/attentive matching -> writes P columns of mv at col0
__global__ void bimpm_mp_full(const float* __restrict__ v1, const float* __restrict__ v2,
                              const float* __restrict__ w, float* __restrict__ mv,
                              int per_l, int fixed_l, int col0) {
  int i = blockIdx.x * blockDim.x + threadIdx.x;   // (b*L+l)*P+p
  if (i >= BLN * PPN) return;
  int p = i % PPN, l = (i / PPN) % LLN, b = i / (PPN * LLN);
  const float* r1 = v1 + (long long)(b * LLN + l) * HHN;
  int l2 = per_l ? l : fixed_l;
  const float* r2 = v2 + (long long)(b * LLN + l2) * HHN;
  const float* wp = w + p * HHN;
  float num = 0.f, n1 = 0.f, n2 = 0.f;
  for (int h = 0; h < HHN; ++h) {
    float a = r1[h] * wp[h], c = r2[h] * wp[h];
    num += a * c; n1 += a * a; n2 += c * c;
  }
  float den = sqrtf(n1) * sqrtf(n2);
  mv[((long long)(b * LLN + l)) * MVW + col0 + p] = num / fmaxf(den, EPS_F);
}

// pairwise maxpool: max over m -> mv_p columns
__global__ void bimpm_pw_max_m(const float* __restrict__ S, const float* __restrict__ wn1,
                               const float* __restrict__ wn2, float* __restrict__ mvp, int col0) {
  int i = blockIdx.x * blockDim.x + threadIdx.x;   // (b*P+p)*L + l
  if (i >= BB * PPN * LLN) return;
  int l = i % LLN, p = (i / LLN) % PPN, b = i / (LLN * PPN);
  const float* Sr = S + (long long)i * LLN;
  float a1 = wn1[i];
  float best = -3.4e38f;
  for (int m = 0; m < LLN; ++m)
    best = fmaxf(best, divsv(Sr[m], a1 * wn2[(b * PPN + p) * LLN + m]));
  mvp[((long long)(b * LLN + l)) * MVW + col0 + p] = best;
}
// pairwise maxpool: max over l -> mv_h columns
__global__ void bimpm_pw_max_l(const float* __restrict__ S, const float* __restrict__ wn1,
                               const float* __restrict__ wn2, float* __restrict__ mvh, int col0) {
  int i = blockIdx.x * blockDim.x + threadIdx.x;   // (b*P+p)*L + m
  if (i >= BB * PPN * LLN) return;
  int m = i % LLN, p = (i / LLN) % PPN, b = i / (LLN * PPN);
  float a2 = wn2[i];
  float best = -3.4e38f;
  for (int l = 0; l < LLN; ++l)
    best = fmaxf(best, divsv(S[((long long)((b * PPN + p) * LLN + l)) * LLN + m],
                             wn1[(b * PPN + p) * LLN + l] * a2));
  mvh[((long long)(b * LLN + m)) * MVW + col0 + p] = best;
}

// ---------------- FC head ---------------------------------------------------
__global__ void bimpm_fc1(const float* pf, const float* pb, const float* hf, const float* hb,
                          const float* __restrict__ W, const float* __restrict__ bias,
                          float* __restrict__ out) {
  int i = blockIdx.x * blockDim.x + threadIdx.x;   // b*256 + j
  if (i >= BB * 256) return;
  int j = i % 256, b = i / 256;
  float acc = bias[j];
  const float* Wr = W + (long long)j * 512;
  for (int k = 0; k < 128; ++k) acc += pf[b * 128 + k] * Wr[k];
  for (int k = 0; k < 128; ++k) acc += pb[b * 128 + k] * Wr[128 + k];
  for (int k = 0; k < 128; ++k) acc += hf[b * 128 + k] * Wr[256 + k];
  for (int k = 0; k < 128; ++k) acc += hb[b * 128 + k] * Wr[384 + k];
  out[i] = tanhf(acc);
}

__global__ void bimpm_fc2(const float* __restrict__ x, const float* __restrict__ W,
                          const float* __restrict__ bias, float* __restrict__ out) {
  int i = blockIdx.x * blockDim.x + threadIdx.x;   // b*2 + j
  if (i >= BB * 2) return;
  int j = i % 2, b = i / 2;
  float acc = bias[j];
  for (int k = 0; k < 256; ++k) acc += x[b * 256 + k] * W[j * 256 + k];
  out[i] = acc;
}

// ---------------- host orchestration ----------------------------------------
extern "C" void kernel_launch(void* const* d_in, const int* in_sizes, int n_in,
                              void* d_out, int out_size, void* d_ws, size_t ws_size,
                              hipStream_t stream) {
  const int*   text_a = (const int*)  d_in[0];
  const int*   text_b = (const int*)  d_in[1];
  const float* emb    = (const float*)d_in[2];
  const float* cWih_f = (const float*)d_in[3];
  const float* cWhh_f = (const float*)d_in[4];
  const float* cbih_f = (const float*)d_in[5];
  const float* cbhh_f = (const float*)d_in[6];
  const float* cWih_b = (const float*)d_in[7];
  const float* cWhh_b = (const float*)d_in[8];
  const float* cbih_b = (const float*)d_in[9];
  const float* cbhh_b = (const float*)d_in[10];
  const float* aWih_f = (const float*)d_in[11];
  const float* aWhh_f = (const float*)d_in[12];
  const float* abih_f = (const float*)d_in[13];
  const float* abhh_f = (const float*)d_in[14];
  const float* aWih_b = (const float*)d_in[15];
  const float* aWhh_b = (const float*)d_in[16];
  const float* abih_b = (const float*)d_in[17];
  const float* abhh_b = (const float*)d_in[18];
  const float* fc1_W  = (const float*)d_in[19];
  const float* fc1_b  = (const float*)d_in[20];
  const float* fc2_W  = (const float*)d_in[21];
  const float* fc2_b  = (const float*)d_in[22];
  const float* mpw[8];
  for (int i = 0; i < 8; ++i) mpw[i] = (const float*)d_in[23 + i];

  char* ws = (char*)d_ws;
  size_t off = 0;
  auto alloc = [&](size_t bytes) -> void* {
    void* p = ws + off;
    off += (bytes + 255) & ~(size_t)255;
    return p;
  };
  auto allocF = [&](size_t n) -> float*     { return (float*)alloc(n * 4); };
  auto allocH = [&](size_t n) -> _Float16*  { return (_Float16*)alloc(n * 2); };

  // f16 staging
  _Float16* xp16 = allocH((size_t)BLN * EPAD);
  _Float16* xh16 = allocH((size_t)BLN * EPAD);
  _Float16* WihF16 = allocH((size_t)GGN * EPAD);
  _Float16* WihB16 = allocH((size_t)GGN * EPAD);
  _Float16* WhhF16 = allocH((size_t)GGN * HHN);
  _Float16* WhhB16 = allocH((size_t)GGN * HHN);
  _Float16* aWihF16 = allocH((size_t)GGN * MVW);
  _Float16* aWihB16 = allocH((size_t)GGN * MVW);
  _Float16* aWhhF16 = allocH((size_t)GGN * HHN);
  _Float16* aWhhB16 = allocH((size_t)GGN * HHN);
  // gate pre-activations
  float* xg_pf = allocF((size_t)BLN * GGN);
  float* xg_pb = allocF((size_t)BLN * GGN);
  float* xg_hf = allocF((size_t)BLN * GGN);
  float* xg_hb = allocF((size_t)BLN * GGN);
  // contextual states
  float* con_p_fw = allocF((size_t)BLN * HHN);
  float* con_p_bw = allocF((size_t)BLN * HHN);
  float* con_h_fw = allocF((size_t)BLN * HHN);
  float* con_h_bw = allocF((size_t)BLN * HHN);
  float* hT_scr = allocF((size_t)4 * BB * HHN);
  _Float16* conpF16 = allocH((size_t)BLN * HHN);
  _Float16* conpB16 = allocH((size_t)BLN * HHN);
  _Float16* conhF16 = allocH((size_t)BLN * HHN);
  _Float16* conhB16 = allocH((size_t)BLN * HHN);
  // attention
  float* S_fw = allocF((size_t)BB * LLN * LLN);
  float* S_bw = allocF((size_t)BB * LLN * LLN);
  float* n_pf = allocF(BLN); float* n_pb = allocF(BLN);
  float* n_hf = allocF(BLN); float* n_hb = allocF(BLN);
  float* rs_fw = allocF(BLN); float* cs_fw = allocF(BLN);
  float* rs_bw = allocF(BLN); float* cs_bw = allocF(BLN);
  _Float16* att16_fw  = allocH((size_t)BB * LLN * LLN);
  _Float16* attT16_fw = allocH((size_t)BB * LLN * LLN);
  _Float16* att16_bw  = allocH((size_t)BB * LLN * LLN);
  _Float16* attT16_bw = allocH((size_t)BB * LLN * LLN);
  float* mean_h_fw = allocF((size_t)BLN * HHN);
  float* mean_h_bw = allocF((size_t)BLN * HHN);
  float* mean_p_fw = allocF((size_t)BLN * HHN);
  float* mean_p_bw = allocF((size_t)BLN * HHN);
  float* max_h_fw = allocF((size_t)BLN * HHN);
  float* max_h_bw = allocF((size_t)BLN * HHN);
  float* max_p_fw = allocF((size_t)BLN * HHN);
  float* max_p_bw = allocF((size_t)BLN * HHN);
  // pairwise
  _Float16* pw_pf = allocH((size_t)BB * PPN * LLN * HHN);
  _Float16* pw_hf = allocH((size_t)BB * PPN * LLN * HHN);
  _Float16* pw_pb = allocH((size_t)BB * PPN * LLN * HHN);
  _Float16* pw_hb = allocH((size_t)BB * PPN * LLN * HHN);
  float* wn_pf = allocF((size_t)BB * PPN * LLN);
  float* wn_hf = allocF((size_t)BB * PPN * LLN);
  float* wn_pb = allocF((size_t)BB * PPN * LLN);
  float* wn_hb = allocF((size_t)BB * PPN * LLN);
  float* Spw_fw = allocF((size_t)BB * PPN * LLN * LLN);
  float* Spw_bw = allocF((size_t)BB * PPN * LLN * LLN);
  // matching vectors + aggregation
  float* mv_p = allocF((size_t)BLN * MVW);
  float* mv_h = allocF((size_t)BLN * MVW);
  _Float16* mvp16 = allocH((size_t)BLN * MVW);
  _Float16* mvh16 = allocH((size_t)BLN * MVW);
  float* xga_pf = allocF((size_t)BLN * GGN);
  float* xga_pb = allocF((size_t)BLN * GGN);
  float* xga_hf = allocF((size_t)BLN * GGN);
  float* xga_hb = allocF((size_t)BLN * GGN);
  float* hs_scr0 = allocF((size_t)BLN * HHN);
  float* hs_scr1 = allocF((size_t)BLN * HHN);
  float* hs_scr2 = allocF((size_t)BLN * HHN);
  float* hs_scr3 = allocF((size_t)BLN * HHN);
  float* p_fwT = allocF(BB * HHN); float* p_bwT = allocF(BB * HHN);
  float* h_fwT = allocF(BB * HHN); float* h_bwT = allocF(BB * HHN);
  float* fc1o = allocF(BB * 256);
  (void)ws_size; (void)in_sizes; (void)n_in; (void)out_size;

  auto cdiv = [](int a, int b) { return (a + b - 1) / b; };

  // 1. embedding gather -> padded f16
  bimpm_gather_embed<<<cdiv(BLN * EPAD, 256), 256, 0, stream>>>(text_a, emb, xp16, BLN * EPAD);
  bimpm_gather_embed<<<cdiv(BLN * EPAD, 256), 256, 0, stream>>>(text_b, emb, xh16, BLN * EPAD);
  // 2. weight conversions
  bimpm_convert_pad<<<cdiv(GGN * EPAD, 256), 256, 0, stream>>>(cWih_f, WihF16, GGN, EE, EPAD);
  bimpm_convert_pad<<<cdiv(GGN * EPAD, 256), 256, 0, stream>>>(cWih_b, WihB16, GGN, EE, EPAD);
  bimpm_convert_pad<<<cdiv(GGN * HHN, 256), 256, 0, stream>>>(cWhh_f, WhhF16, GGN, HHN, HHN);
  bimpm_convert_pad<<<cdiv(GGN * HHN, 256), 256, 0, stream>>>(cWhh_b, WhhB16, GGN, HHN, HHN);
  bimpm_convert_pad<<<cdiv(GGN * MVW, 256), 256, 0, stream>>>(aWih_f, aWihF16, GGN, MVW, MVW);
  bimpm_convert_pad<<<cdiv(GGN * MVW, 256), 256, 0, stream>>>(aWih_b, aWihB16, GGN, MVW, MVW);
  bimpm_convert_pad<<<cdiv(GGN * HHN, 256), 256, 0, stream>>>(aWhh_f, aWhhF16, GGN, HHN, HHN);
  bimpm_convert_pad<<<cdiv(GGN * HHN, 256), 256, 0, stream>>>(aWhh_b, aWhhB16, GGN, HHN, HHN);
  // 3. input projections (WMMA): xg = x @ Wih^T + biases
  {
    int tiles = (BLN / 16) * (GGN / 16);
    dim3 g(cdiv(tiles, 4), 1);
    bimpm_gemm_nt<<<g, 128, 0, stream>>>(xp16, WihF16, xg_pf, BLN, GGN, EPAD, 0, 0, 0);
    bimpm_gemm_nt<<<g, 128, 0, stream>>>(xp16, WihB16, xg_pb, BLN, GGN, EPAD, 0, 0, 0);
    bimpm_gemm_nt<<<g, 128, 0, stream>>>(xh16, WihF16, xg_hf, BLN, GGN, EPAD, 0, 0, 0);
    bimpm_gemm_nt<<<g, 128, 0, stream>>>(xh16, WihB16, xg_hb, BLN, GGN, EPAD, 0, 0, 0);
  }
  int bt = BLN * GGN;
  bimpm_add_bias2<<<cdiv(bt, 256), 256, 0, stream>>>(xg_pf, cbih_f, cbhh_f, bt);
  bimpm_add_bias2<<<cdiv(bt, 256), 256, 0, stream>>>(xg_pb, cbih_b, cbhh_b, bt);
  bimpm_add_bias2<<<cdiv(bt, 256), 256, 0, stream>>>(xg_hf, cbih_f, cbhh_f, bt);
  bimpm_add_bias2<<<cdiv(bt, 256), 256, 0, stream>>>(xg_hb, cbih_b, cbhh_b, bt);
  // 4. context BiLSTM scans (WMMA recurrence)
  bimpm_lstm_scan4<<<4, 256, 0, stream>>>(xg_pf, xg_hf, xg_pb, xg_hb, WhhF16, WhhB16,
      con_p_fw, con_h_fw, con_p_bw, con_h_bw,
      hT_scr, hT_scr + BB*HHN, hT_scr + 2*BB*HHN, hT_scr + 3*BB*HHN);
  // 5. f16 copies of contextual states
  int cn = BLN * HHN;
  bimpm_f32_to_f16<<<cdiv(cn, 256), 256, 0, stream>>>(con_p_fw, conpF16, cn);
  bimpm_f32_to_f16<<<cdiv(cn, 256), 256, 0, stream>>>(con_p_bw, conpB16, cn);
  bimpm_f32_to_f16<<<cdiv(cn, 256), 256, 0, stream>>>(con_h_fw, conhF16, cn);
  bimpm_f32_to_f16<<<cdiv(cn, 256), 256, 0, stream>>>(con_h_bw, conhB16, cn);
  // 6. attention numerators (batched WMMA, NT): S[b] = conp[b] @ conh[b]^T
  {
    dim3 g(cdiv((LLN/16)*(LLN/16), 4), BB);
    bimpm_gemm_nt<<<g, 128, 0, stream>>>(conpF16, conhF16, S_fw, LLN, LLN, HHN,
                                         LLN*HHN, LLN*HHN, LLN*LLN);
    bimpm_gemm_nt<<<g, 128, 0, stream>>>(conpB16, conhB16, S_bw, LLN, LLN, HHN,
                                         LLN*HHN, LLN*HHN, LLN*LLN);
  }
  // 7. norms + normalize attention
  bimpm_norm<<<cdiv(BLN, 256), 256, 0, stream>>>(con_p_fw, n_pf);
  bimpm_norm<<<cdiv(BLN, 256), 256, 0, stream>>>(con_p_bw, n_pb);
  bimpm_norm<<<cdiv(BLN, 256), 256, 0, stream>>>(con_h_fw, n_hf);
  bimpm_norm<<<cdiv(BLN, 256), 256, 0, stream>>>(con_h_bw, n_hb);
  int an = BB * LLN * LLN;
  bimpm_att_normalize<<<cdiv(an, 256), 256, 0, stream>>>(S_fw, n_pf, n_hf);
  bimpm_att_normalize<<<cdiv(an, 256), 256, 0, stream>>>(S_bw, n_pb, n_hb);
  bimpm_rowsum<<<cdiv(BLN, 256), 256, 0, stream>>>(S_fw, rs_fw);
  bimpm_colsum<<<cdiv(BLN, 256), 256, 0, stream>>>(S_fw, cs_fw);
  bimpm_rowsum<<<cdiv(BLN, 256), 256, 0, stream>>>(S_bw, rs_bw);
  bimpm_colsum<<<cdiv(BLN, 256), 256, 0, stream>>>(S_bw, cs_bw);
  bimpm_att_f16_tr<<<cdiv(an, 256), 256, 0, stream>>>(S_fw, att16_fw, attT16_fw);
  bimpm_att_f16_tr<<<cdiv(an, 256), 256, 0, stream>>>(S_bw, att16_bw, attT16_bw);
  // 8. attentive means (batched WMMA, NN) then divide by att sums
  {
    dim3 g(cdiv((LLN/16)*(HHN/16), 4), BB);
    bimpm_gemm_nn<<<g, 128, 0, stream>>>(att16_fw,  conhF16, mean_h_fw, LLN, HHN, LLN,
                                         LLN*LLN, LLN*HHN, LLN*HHN);
    bimpm_gemm_nn<<<g, 128, 0, stream>>>(attT16_fw, conpF16, mean_p_fw, LLN, HHN, LLN,
                                         LLN*LLN, LLN*HHN, LLN*HHN);
    bimpm_gemm_nn<<<g, 128, 0, stream>>>(att16_bw,  conhB16, mean_h_bw, LLN, HHN, LLN,
                                         LLN*LLN, LLN*HHN, LLN*HHN);
    bimpm_gemm_nn<<<g, 128, 0, stream>>>(attT16_bw, conpB16, mean_p_bw, LLN, HHN, LLN,
                                         LLN*LLN, LLN*HHN, LLN*HHN);
  }
  bimpm_div_rows<<<cdiv(cn, 256), 256, 0, stream>>>(mean_h_fw, rs_fw, cn);
  bimpm_div_rows<<<cdiv(cn, 256), 256, 0, stream>>>(mean_p_fw, cs_fw, cn);
  bimpm_div_rows<<<cdiv(cn, 256), 256, 0, stream>>>(mean_h_bw, rs_bw, cn);
  bimpm_div_rows<<<cdiv(cn, 256), 256, 0, stream>>>(mean_p_bw, cs_bw, cn);
  // 9. attentive maxes
  bimpm_attmax_h<<<cdiv(cn, 256), 256, 0, stream>>>(S_fw, con_h_fw, max_h_fw);
  bimpm_attmax_p<<<cdiv(cn, 256), 256, 0, stream>>>(S_fw, con_p_fw, max_p_fw);
  bimpm_attmax_h<<<cdiv(cn, 256), 256, 0, stream>>>(S_bw, con_h_bw, max_h_bw);
  bimpm_attmax_p<<<cdiv(cn, 256), 256, 0, stream>>>(S_bw, con_p_bw, max_p_bw);
  // 10. pairwise maxpool matching (w3 fw / w4 bw) via batched WMMA
  int pn = BB * PPN * LLN * HHN;
  bimpm_scale_w_f16<<<cdiv(pn, 256), 256, 0, stream>>>(con_p_fw, mpw[2], pw_pf);
  bimpm_scale_w_f16<<<cdiv(pn, 256), 256, 0, stream>>>(con_h_fw, mpw[2], pw_hf);
  bimpm_scale_w_f16<<<cdiv(pn, 256), 256, 0, stream>>>(con_p_bw, mpw[3], pw_pb);
  bimpm_scale_w_f16<<<cdiv(pn, 256), 256, 0, stream>>>(con_h_bw, mpw[3], pw_hb);
  int wnN = BB * PPN * LLN;
  bimpm_wnorm<<<cdiv(wnN, 256), 256, 0, stream>>>(con_p_fw, mpw[2], wn_pf);
  bimpm_wnorm<<<cdiv(wnN, 256), 256, 0, stream>>>(con_h_fw, mpw[2], wn_hf);
  bimpm_wnorm<<<cdiv(wnN, 256), 256, 0, stream>>>(con_p_bw, mpw[3], wn_pb);
  bimpm_wnorm<<<cdiv(wnN, 256), 256, 0, stream>>>(con_h_bw, mpw[3], wn_hb);
  {
    dim3 g(cdiv((LLN/16)*(LLN/16), 4), BB * PPN);
    bimpm_gemm_nt<<<g, 128, 0, stream>>>(pw_pf, pw_hf, Spw_fw, LLN, LLN, HHN,
                                         LLN*HHN, LLN*HHN, LLN*LLN);
    bimpm_gemm_nt<<<g, 128, 0, stream>>>(pw_pb, pw_hb, Spw_bw, LLN, LLN, HHN,
                                         LLN*HHN, LLN*HHN, LLN*LLN);
  }
  bimpm_pw_max_m<<<cdiv(wnN, 256), 256, 0, stream>>>(Spw_fw, wn_pf, wn_hf, mv_p, 4);
  bimpm_pw_max_l<<<cdiv(wnN, 256), 256, 0, stream>>>(Spw_fw, wn_pf, wn_hf, mv_h, 4);
  bimpm_pw_max_m<<<cdiv(wnN, 256), 256, 0, stream>>>(Spw_bw, wn_pb, wn_hb, mv_p, 20);
  bimpm_pw_max_l<<<cdiv(wnN, 256), 256, 0, stream>>>(Spw_bw, wn_pb, wn_hb, mv_h, 20);
  // 11. full + attentive matching columns
  int fn = BLN * PPN, fg = cdiv(fn, 256);
  bimpm_mp_full<<<fg, 256, 0, stream>>>(con_p_fw, con_h_fw, mpw[0], mv_p, 0, LLN - 1, 0);
  bimpm_mp_full<<<fg, 256, 0, stream>>>(con_p_bw, con_h_bw, mpw[1], mv_p, 0, 0, 16);
  bimpm_mp_full<<<fg, 256, 0, stream>>>(con_p_fw, mean_h_fw, mpw[4], mv_p, 1, 0, 8);
  bimpm_mp_full<<<fg, 256, 0, stream>>>(con_p_bw, mean_h_bw, mpw[5], mv_p, 1, 0, 24);
  bimpm_mp_full<<<fg, 256, 0, stream>>>(con_p_fw, max_h_fw, mpw[6], mv_p, 1, 0, 12);
  bimpm_mp_full<<<fg, 256, 0, stream>>>(con_p_bw, max_h_bw, mpw[7], mv_p, 1, 0, 28);
  bimpm_mp_full<<<fg, 256, 0, stream>>>(con_h_fw, con_p_fw, mpw[0], mv_h, 0, LLN - 1, 0);
  bimpm_mp_full<<<fg, 256, 0, stream>>>(con_h_bw, con_p_bw, mpw[1], mv_h, 0, 0, 16);
  bimpm_mp_full<<<fg, 256, 0, stream>>>(con_h_fw, mean_p_fw, mpw[4], mv_h, 1, 0, 8);
  bimpm_mp_full<<<fg, 256, 0, stream>>>(con_h_bw, mean_p_bw, mpw[5], mv_h, 1, 0, 24);
  bimpm_mp_full<<<fg, 256, 0, stream>>>(con_h_fw, max_p_fw, mpw[6], mv_h, 1, 0, 12);
  bimpm_mp_full<<<fg, 256, 0, stream>>>(con_h_bw, max_p_bw, mpw[7], mv_h, 1, 0, 28);
  // 12. aggregation BiLSTM
  int mn = BLN * MVW;
  bimpm_f32_to_f16<<<cdiv(mn, 256), 256, 0, stream>>>(mv_p, mvp16, mn);
  bimpm_f32_to_f16<<<cdiv(mn, 256), 256, 0, stream>>>(mv_h, mvh16, mn);
  {
    int tiles = (BLN / 16) * (GGN / 16);
    dim3 g(cdiv(tiles, 4), 1);
    bimpm_gemm_nt<<<g, 128, 0, stream>>>(mvp16, aWihF16, xga_pf, BLN, GGN, MVW, 0, 0, 0);
    bimpm_gemm_nt<<<g, 128, 0, stream>>>(mvp16, aWihB16, xga_pb, BLN, GGN, MVW, 0, 0, 0);
    bimpm_gemm_nt<<<g, 128, 0, stream>>>(mvh16, aWihF16, xga_hf, BLN, GGN, MVW, 0, 0, 0);
    bimpm_gemm_nt<<<g, 128, 0, stream>>>(mvh16, aWihB16, xga_hb, BLN, GGN, MVW, 0, 0, 0);
  }
  bimpm_add_bias2<<<cdiv(bt, 256), 256, 0, stream>>>(xga_pf, abih_f, abhh_f, bt);
  bimpm_add_bias2<<<cdiv(bt, 256), 256, 0, stream>>>(xga_pb, abih_b, abhh_b, bt);
  bimpm_add_bias2<<<cdiv(bt, 256), 256, 0, stream>>>(xga_hf, abih_f, abhh_f, bt);
  bimpm_add_bias2<<<cdiv(bt, 256), 256, 0, stream>>>(xga_hb, abih_b, abhh_b, bt);
  bimpm_lstm_scan4<<<4, 256, 0, stream>>>(xga_pf, xga_hf, xga_pb, xga_hb, aWhhF16, aWhhB16,
      hs_scr0, hs_scr1, hs_scr2, hs_scr3, p_fwT, h_fwT, p_bwT, h_bwT);
  // 13. FC head
  bimpm_fc1<<<cdiv(BB * 256, 256), 256, 0, stream>>>(p_fwT, p_bwT, h_fwT, h_bwT,
                                                     fc1_W, fc1_b, fc1o);
  bimpm_fc2<<<1, 64, 0, stream>>>(fc1o, fc2_W, fc2_b, (float*)d_out);
}
